// GraphTransformer_19069654794757
// MI455X (gfx1250) — compile-verified, matrix operand order checked by reading, not run on hardware
//
#include <hip/hip_runtime.h>
#include <math.h>

#define NN_ 16384
#define EE_ 131072
#define DD_ 128
#define HH_ 4
#define LL_ 4
#define GG_ 64
#define SS_ 256

#ifndef __has_builtin
#define __has_builtin(x) 0
#endif

#if __has_builtin(__builtin_amdgcn_tensor_load_to_lds) && \
    __has_builtin(__builtin_amdgcn_s_wait_tensorcnt)
#define HAVE_TDM 1
#if defined(__has_include)
#if __has_include(<hip/amd_detail/amd_gfx1250_TDM.h>)
#define TDM_ARGS6 1
#endif
#endif
#else
#define HAVE_TDM 0
#endif

typedef __attribute__((ext_vector_type(16))) __bf16 v16bf;
typedef __attribute__((ext_vector_type(8)))  float v8f;
typedef __attribute__((ext_vector_type(4)))  unsigned int u32x4;
typedef __attribute__((ext_vector_type(8)))  int i32x8;
typedef __attribute__((ext_vector_type(4)))  int i32x4;

__device__ __forceinline__ unsigned short f2bf(float f) {
    unsigned int u = __float_as_uint(f);
    return (unsigned short)((u + 0x7FFFu + ((u >> 16) & 1u)) >> 16);
}
__device__ __forceinline__ float bf2f(unsigned short h) {
    return __uint_as_float(((unsigned int)h) << 16);
}

#if HAVE_TDM
// Issue one TDM 2D tile load: global [rows x 128 halfs] (row pitch = ld elems)
// -> LDS contiguous rows of 128 halfs. Per CDNA5 ISA D# layout.
__device__ __forceinline__ void tdm_load_2d(unsigned lds_off,
                                            const unsigned short* gptr,
                                            long long ld, int rows) {
    unsigned long long ga = (unsigned long long)gptr;
    const unsigned tk = 128u;  // tile/tensor dim0 (elements)
    u32x4 g0;
    g0[0] = 1u;                                      // count=1, user descriptor
    g0[1] = lds_off;                                 // lds_addr (bytes)
    g0[2] = (unsigned)(ga & 0xFFFFFFFFull);          // global_addr[31:0]
    g0[3] = (unsigned)((ga >> 32) & 0x01FFFFFFull) | 0x80000000u;  // [56:32] | type=2
    unsigned long long st = (unsigned long long)ld;  // dim0 stride (elements)
    i32x8 g1;
    g1[0] = 0x00010000;                              // data_size=1 -> 2 bytes
    g1[1] = (int)(tk << 16);                         // tensor_dim0[15:0] @ 63:48
    g1[2] = (int)((tk >> 16) | ((unsigned)rows << 16));      // tdim0 hi | tdim1 lo
    g1[3] = (int)(((unsigned)rows >> 16) | (tk << 16));      // tdim1 hi | tile_dim0
    g1[4] = (int)(unsigned)rows;                     // tile_dim1 | tile_dim2=0
    g1[5] = (int)(unsigned)(st & 0xFFFFFFFFull);     // dim0 stride [31:0]
    g1[6] = (int)(unsigned)((st >> 32) & 0xFFFFull); // stride [47:32] | dim1stride lo=0
    g1[7] = 0;
    i32x4 z4 = {};
#ifdef TDM_ARGS6
    i32x8 z8 = {};
    __builtin_amdgcn_tensor_load_to_lds(g0, g1, z4, z4, z8, 0);
#else
    __builtin_amdgcn_tensor_load_to_lds(g0, g1, z4, z4, 0);
#endif
}
#endif

// ---------------------------------------------------------------------------
// bf16 WMMA GEMM: C[M,Nout] = A[M,K] @ op(B) + bias, opt ReLU.
// Block tile 128x64 (48 KB LDS), 8 waves as 4(M)x2(N); each wave owns a 32x32
// tile = 2x2 WMMA accumulators (4 wmma per 8 ds_load_b128 per K-step).
// transB=1: B stored [Nout,K] row-major (A @ B^T) -> both tiles TDM-loadable.
// gridDim.z batches independent problems via element strides.
// ---------------------------------------------------------------------------
__global__ __launch_bounds__(256) void gemm_wmma_kernel(
    const unsigned short* __restrict__ A, long long lda, long long sAg,
    const unsigned short* __restrict__ B, long long ldb, long long sBg,
    const float* __restrict__ bias,
    float* __restrict__ C, unsigned short* __restrict__ Cb,
    long long ldc, long long sCg,
    int K, int transB, int relu)
{
    __shared__ unsigned short As[128 * 128];  // [m][k] 32 KB
    __shared__ unsigned short Bs[64 * 128];   // [n][k] 16 KB

    const int tid  = threadIdx.x;
    const int lane = tid & 31;
    const int wave = tid >> 5;
    const int wm   = (wave & 3) * 32;   // wave M offset in block tile
    const int wn   = (wave >> 2) * 32;  // wave N offset in block tile
    const long long bm = (long long)blockIdx.y * 128;
    const long long bn = (long long)blockIdx.x * 64;
    const int gz = blockIdx.z;

    A += (long long)gz * sAg;
    B += (long long)gz * sBg;

    union Frag { v16bf v; unsigned int u[8]; };
    v8f acc[2][2] = {};

    const int g  = lane >> 4;   // K-half select
    const int lm = lane & 15;   // row (A) / col (B) within 16

#if HAVE_TDM
    const unsigned asOff = (unsigned)(unsigned long long)(const void*)As;
    const unsigned bsOff = (unsigned)(unsigned long long)(const void*)Bs;
#endif

    for (int kc = 0; kc < K; kc += 128) {
        __syncthreads();
#if HAVE_TDM
        if (wave == 0) {
            tdm_load_2d(asOff, &A[bm * lda + kc], lda, 128);
            if (transB) tdm_load_2d(bsOff, &B[bn * ldb + kc], ldb, 64);
        }
#else
        for (int idx = tid; idx < 128 * 16; idx += 256) {
            int m = idx >> 4, seg = idx & 15;
            *(uint4*)(&As[m * 128 + seg * 8]) =
                *(const uint4*)(&A[(bm + m) * lda + kc + seg * 8]);
        }
        if (transB) {
            for (int idx = tid; idx < 64 * 16; idx += 256) {
                int n = idx >> 4, seg = idx & 15;
                *(uint4*)(&Bs[n * 128 + seg * 8]) =
                    *(const uint4*)(&B[(bn + n) * ldb + kc + seg * 8]);
            }
        }
#endif
        if (!transB) {
            // transpose-stage B [K][Nout] -> Bs[n][k]
            for (int idx = tid; idx < 128 * 8; idx += 256) {
                int krow = idx >> 3, seg = idx & 7;
                union { uint4 q; unsigned short h[8]; } u;
                u.q = *(const uint4*)(&B[(long long)(kc + krow) * ldb + bn + seg * 8]);
#pragma unroll
                for (int j = 0; j < 8; ++j)
                    Bs[(seg * 8 + j) * 128 + krow] = u.h[j];
            }
        }
#if HAVE_TDM
        if (wave == 0) __builtin_amdgcn_s_wait_tensorcnt(0);
#endif
        __syncthreads();

#pragma unroll
        for (int ks = 0; ks < 4; ++ks) {
            Frag fa[2], fb[2];
#pragma unroll
            for (int mi = 0; mi < 2; ++mi) {
                const unsigned int* Ap =
                    (const unsigned int*)(&As[(wm + mi * 16 + lm) * 128 + ks * 32]);
#pragma unroll
                for (int j = 0; j < 8; ++j)
                    fa[mi].u[j] = Ap[j + ((j >= 4) ? 4 : 0) + 4 * g];
            }
#pragma unroll
            for (int ni = 0; ni < 2; ++ni) {
                const unsigned int* Bp =
                    (const unsigned int*)(&Bs[(wn + ni * 16 + lm) * 128 + ks * 32]);
#pragma unroll
                for (int j = 0; j < 8; ++j)
                    fb[ni].u[j] = Bp[j + ((j >= 4) ? 4 : 0) + 4 * g];
            }
#pragma unroll
            for (int mi = 0; mi < 2; ++mi)
#pragma unroll
                for (int ni = 0; ni < 2; ++ni)
                    acc[mi][ni] = __builtin_amdgcn_wmma_f32_16x16x32_bf16(
                        false, fa[mi].v, false, fb[ni].v, (short)0,
                        acc[mi][ni], false, false);
        }
    }

#pragma unroll
    for (int mi = 0; mi < 2; ++mi) {
#pragma unroll
        for (int ni = 0; ni < 2; ++ni) {
            const long long col = bn + wn + ni * 16 + lm;
            const float bvv = bias ? bias[col] : 0.0f;
#pragma unroll
            for (int r = 0; r < 8; ++r) {
                long long row = bm + wm + mi * 16 + r + 8 * g;
                float val = acc[mi][ni][r] + bvv;
                if (relu) val = fmaxf(val, 0.0f);
                long long off = (long long)gz * sCg + row * ldc + col;
                if (C)  C[off]  = val;
                if (Cb) Cb[off] = f2bf(val);
            }
        }
    }
}

// ---------------------------------------------------------------------------
// Elementwise / encoder / reduction kernels
// ---------------------------------------------------------------------------
__global__ void convert_f32_bf16_kernel(const float* __restrict__ in,
                                        unsigned short* __restrict__ out, long long n) {
    long long i = (long long)blockIdx.x * 256 + threadIdx.x;
    if (i < n) out[i] = f2bf(in[i]);
}

// convert + transpose weights [L][K][Nout] fp32 -> [L][Nout][K] bf16
__global__ void wtrans_kernel(const float* __restrict__ W,
                              unsigned short* __restrict__ Wt,
                              int K, int Nout, long long total) {
    long long i = (long long)blockIdx.x * 256 + threadIdx.x;
    if (i >= total) return;
    long long per = (long long)K * Nout;
    int l = (int)(i / per);
    long long r = i - (long long)l * per;
    int k = (int)(r / Nout);
    int n = (int)(r - (long long)k * Nout);
    Wt[(long long)l * per + (long long)n * K + k] = f2bf(W[i]);
}

__global__ void fill_kernel(float* __restrict__ p, float v, long long n) {
    long long i = (long long)blockIdx.x * 256 + threadIdx.x;
    if (i < n) p[i] = v;
}

__global__ void add2_kernel(const float* __restrict__ a, const float* __restrict__ b,
                            float* __restrict__ o, long long n) {
    long long i = (long long)blockIdx.x * 256 + threadIdx.x;
    if (i < n) o[i] = a[i] + b[i];
}

__global__ void atom_encode_kernel(const int* __restrict__ xf,
                                   const float* __restrict__ emb,
                                   const float* __restrict__ pe,
                                   const float* __restrict__ Wpe,
                                   const float* __restrict__ bpe,
                                   float* __restrict__ x,
                                   unsigned short* __restrict__ xb) {
    long long i = (long long)blockIdx.x * 256 + threadIdx.x;
    if (i >= (long long)NN_ * DD_) return;
    int n = (int)(i >> 7), d = (int)(i & 127);
    float s = bpe[d];
#pragma unroll
    for (int f = 0; f < 9; ++f) {
        int val = xf[n * 9 + f];
        s += emb[((long long)f * 128 + val) * DD_ + d];
    }
#pragma unroll
    for (int p = 0; p < 16; ++p) s += pe[n * 16 + p] * Wpe[p * DD_ + d];
    x[i] = s;
    xb[i] = f2bf(s);
}

__global__ void bond_encode_kernel(const int* __restrict__ ef,
                                   const float* __restrict__ emb,
                                   unsigned short* __restrict__ eb) {
    long long i = (long long)blockIdx.x * 256 + threadIdx.x;
    if (i >= (long long)EE_ * DD_) return;
    int e = (int)(i >> 7), d = (int)(i & 127);
    float s = 0.0f;
#pragma unroll
    for (int f = 0; f < 3; ++f) {
        int val = ef[e * 3 + f];
        s += emb[((long long)f * 8 + val) * DD_ + d];
    }
    eb[i] = f2bf(s);
}

__device__ __forceinline__ float atomicMaxFloat(float* addr, float value) {
    int* ia = (int*)addr;
    int old = *ia;
    while (__int_as_float(old) < value) {
        int assumed = old;
        old = atomicCAS(ia, assumed, __float_as_int(value));
        if (old == assumed) break;
    }
    return __int_as_float(old);
}

__global__ void edge_logits_kernel(const int* __restrict__ eidx,
                                   const float* __restrict__ q,
                                   const float* __restrict__ k,
                                   const unsigned short* __restrict__ eeb,
                                   float* __restrict__ logits,
                                   float* __restrict__ mbuf) {
    long long i = (long long)blockIdx.x * 256 + threadIdx.x;
    if (i >= (long long)EE_ * HH_) return;
    int e = (int)(i >> 2), h = (int)(i & 3);
    int src = eidx[e], dst = eidx[EE_ + e];
    const float* qp = q + (long long)dst * 512 + h * 128;
    const float* kp = k + (long long)src * 512 + h * 128;
    const unsigned short* ep = eeb + (long long)e * 512 + h * 128;
    float acc = 0.0f;
    for (int d = 0; d < 128; ++d) acc += qp[d] * (kp[d] + bf2f(ep[d]));
    acc *= 0.08838834764831845f;  // 1/sqrt(128)
    logits[i] = acc;
    atomicMaxFloat(&mbuf[(long long)dst * 4 + h], acc);
}

__global__ void edge_exp_sum_kernel(const int* __restrict__ eidx,
                                    float* __restrict__ logits,
                                    const float* __restrict__ mbuf,
                                    float* __restrict__ zbuf) {
    long long i = (long long)blockIdx.x * 256 + threadIdx.x;
    if (i >= (long long)EE_ * HH_) return;
    int e = (int)(i >> 2), h = (int)(i & 3);
    int dst = eidx[EE_ + e];
    float a = expf(logits[i] - mbuf[(long long)dst * 4 + h]);
    logits[i] = a;
    atomicAdd(&zbuf[(long long)dst * 4 + h], a);
}

__global__ void edge_accum_kernel(const int* __restrict__ eidx,
                                  const float* __restrict__ logits,
                                  const float* __restrict__ zbuf,
                                  const float* __restrict__ v,
                                  const unsigned short* __restrict__ eeb,
                                  float* __restrict__ msg) {
    long long i = (long long)blockIdx.x * 256 + threadIdx.x;
    if (i >= (long long)EE_ * 512) return;
    int e = (int)(i >> 9), r = (int)(i & 511);
    int h = r >> 7;
    int src = eidx[e], dst = eidx[EE_ + e];
    float alpha = logits[(long long)e * 4 + h] / zbuf[(long long)dst * 4 + h];
    float vj = v[(long long)src * 512 + r] + bf2f(eeb[(long long)e * 512 + r]);
    atomicAdd(&msg[(long long)dst * 512 + r], alpha * vj);
}

__global__ void combine_local_kernel(const float* __restrict__ msg,
                                     const float* __restrict__ skip,
                                     const float* __restrict__ x,
                                     float* __restrict__ out) {
    long long i = (long long)blockIdx.x * 256 + threadIdx.x;
    if (i >= (long long)NN_ * DD_) return;
    int n = (int)(i >> 7), d = (int)(i & 127);
    const float* mp = msg + (long long)n * 512 + d;
    float mm = 0.25f * (mp[0] + mp[128] + mp[256] + mp[384]);
    out[i] = mm + skip[i] + x[i];
}

__global__ void bn_kernel(const float* __restrict__ in, float* __restrict__ out,
                          unsigned short* __restrict__ outb,
                          const float* __restrict__ gamma,
                          const float* __restrict__ beta, int rows) {
    const int d = blockIdx.x;
    __shared__ float ssum[256], ssq[256];
    float s = 0.0f, q = 0.0f;
    for (int n = threadIdx.x; n < rows; n += 256) {
        float v = in[(long long)n * DD_ + d];
        s += v; q += v * v;
    }
    ssum[threadIdx.x] = s; ssq[threadIdx.x] = q;
    __syncthreads();
    for (int st = 128; st > 0; st >>= 1) {
        if (threadIdx.x < st) {
            ssum[threadIdx.x] += ssum[threadIdx.x + st];
            ssq[threadIdx.x]  += ssq[threadIdx.x + st];
        }
        __syncthreads();
    }
    float mu = ssum[0] / rows;
    float var = ssq[0] / rows - mu * mu;
    float rstd = rsqrtf(var + 1e-5f);
    float gg = gamma[d], bb = beta[d];
    for (int n = threadIdx.x; n < rows; n += 256) {
        long long off = (long long)n * DD_ + d;
        float v = (in[off] - mu) * rstd * gg + bb;
        out[off] = v;
        if (outb) outb[off] = f2bf(v);
    }
}

__global__ void softmax_rows_kernel(const float* __restrict__ in,
                                    unsigned short* __restrict__ outb, float scale) {
    long long row = blockIdx.x;
    __shared__ float red[256];
    int t = threadIdx.x;
    float v = in[row * 256 + t] * scale;
    red[t] = v; __syncthreads();
    for (int st = 128; st > 0; st >>= 1) {
        if (t < st) red[t] = fmaxf(red[t], red[t + st]);
        __syncthreads();
    }
    float mx = red[0]; __syncthreads();
    float e = expf(v - mx);
    red[t] = e; __syncthreads();
    for (int st = 128; st > 0; st >>= 1) {
        if (t < st) red[t] += red[t + st];
        __syncthreads();
    }
    outb[row * 256 + t] = f2bf(e / red[0]);
}

__global__ void head_kernel(const float* __restrict__ x,
                            const float* __restrict__ Wh1, const float* __restrict__ bh1,
                            const float* __restrict__ Wh2, const float* __restrict__ bh2,
                            const float* __restrict__ Wh3, const float* __restrict__ bh3,
                            float* __restrict__ out) {
    int g = blockIdx.x, t = threadIdx.x;  // 128 threads
    __shared__ float pooled[128], h1[64], h2[32];
    float s = 0.0f;
    for (int ss = 0; ss < SS_; ++ss)
        s += x[((long long)g * SS_ + ss) * DD_ + t];
    pooled[t] = s * (1.0f / 256.0f);
    __syncthreads();
    if (t < 64) {
        float a = bh1[t];
        for (int d = 0; d < 128; ++d) a += pooled[d] * Wh1[d * 64 + t];
        h1[t] = fmaxf(a, 0.0f);
    }
    __syncthreads();
    if (t < 32) {
        float a = bh2[t];
        for (int d = 0; d < 64; ++d) a += h1[d] * Wh2[d * 32 + t];
        h2[t] = fmaxf(a, 0.0f);
    }
    __syncthreads();
    if (t == 0) {
        float a = bh3[0];
        for (int d = 0; d < 32; ++d) a += h2[d] * Wh3[d];
        out[g] = a;
    }
}

// ---------------------------------------------------------------------------
// Host side
// ---------------------------------------------------------------------------
static inline char* bump(char*& p, size_t bytes) {
    char* r = p;
    p += (bytes + 255) & ~(size_t)255;
    return r;
}

static void launch_gemm(hipStream_t stream,
                        const unsigned short* A, long long lda, long long sAg,
                        const unsigned short* B, long long ldb, long long sBg,
                        const float* bias, float* C, unsigned short* Cb,
                        long long ldc, long long sCg,
                        int M, int Nout, int K, int transB, int relu, int nG) {
    dim3 grid(Nout / 64, M / 128, nG);
    gemm_wmma_kernel<<<grid, 256, 0, stream>>>(A, lda, sAg, B, ldb, sBg, bias,
                                               C, Cb, ldc, sCg, K, transB, relu);
}

static void cvt(hipStream_t stream, const float* in, unsigned short* out, long long n) {
    convert_f32_bf16_kernel<<<(unsigned)((n + 255) / 256), 256, 0, stream>>>(in, out, n);
}

static void cvtT(hipStream_t stream, const float* W, unsigned short* Wt, int K, int Nout) {
    long long total = (long long)LL_ * K * Nout;
    wtrans_kernel<<<(unsigned)((total + 255) / 256), 256, 0, stream>>>(W, Wt, K, Nout, total);
}

extern "C" void kernel_launch(void* const* d_in, const int* in_sizes, int n_in,
                              void* d_out, int out_size, void* d_ws, size_t ws_size,
                              hipStream_t stream) {
    (void)in_sizes; (void)n_in; (void)out_size; (void)ws_size;
    const int*   x_feat    = (const int*)d_in[0];
    const int*   edge_feat = (const int*)d_in[1];
    const int*   edge_idx  = (const int*)d_in[2];
    const float* pe        = (const float*)d_in[4];
    const float* atom_emb  = (const float*)d_in[5];
    const float* bond_emb  = (const float*)d_in[6];
    const float* Wpe       = (const float*)d_in[7];
    const float* bpe       = (const float*)d_in[8];
    const float* Wq  = (const float*)d_in[9];   const float* bq   = (const float*)d_in[10];
    const float* Wk  = (const float*)d_in[11];  const float* bk   = (const float*)d_in[12];
    const float* Wv  = (const float*)d_in[13];  const float* bv   = (const float*)d_in[14];
    const float* We  = (const float*)d_in[15];
    const float* Wsk = (const float*)d_in[16];  const float* bsk  = (const float*)d_in[17];
    const float* Wqkv= (const float*)d_in[18];  const float* bqkv = (const float*)d_in[19];
    const float* Wo  = (const float*)d_in[20];  const float* bo   = (const float*)d_in[21];
    const float* W1  = (const float*)d_in[22];  const float* b1m  = (const float*)d_in[23];
    const float* W2  = (const float*)d_in[24];  const float* b2m  = (const float*)d_in[25];
    const float* g1  = (const float*)d_in[26];  const float* b1   = (const float*)d_in[27];
    const float* g2  = (const float*)d_in[28];  const float* b2   = (const float*)d_in[29];
    const float* g3  = (const float*)d_in[30];  const float* b3   = (const float*)d_in[31];
    const float* Wh1 = (const float*)d_in[32];  const float* bh1  = (const float*)d_in[33];
    const float* Wh2 = (const float*)d_in[34];  const float* bh2  = (const float*)d_in[35];
    const float* Wh3 = (const float*)d_in[36];  const float* bh3  = (const float*)d_in[37];
    float* out = (float*)d_out;

    char* p = (char*)d_ws;
    float*          x     = (float*)bump(p, (size_t)NN_ * 128 * 4);
    unsigned short* xb    = (unsigned short*)bump(p, (size_t)NN_ * 128 * 2);
    unsigned short* eb    = (unsigned short*)bump(p, (size_t)EE_ * 128 * 2);
    float*          q     = (float*)bump(p, (size_t)NN_ * 512 * 4);
    float*          k     = (float*)bump(p, (size_t)NN_ * 512 * 4);
    float*          v     = (float*)bump(p, (size_t)NN_ * 512 * 4);
    unsigned short* eebuf = (unsigned short*)bump(p, (size_t)EE_ * 512 * 2);
    float*          logit = (float*)bump(p, (size_t)EE_ * 4 * 4);
    float*          mbuf  = (float*)bump(p, (size_t)NN_ * 4 * 4);
    float*          zbuf  = (float*)bump(p, (size_t)NN_ * 4 * 4);
    float*          msg   = (float*)bump(p, (size_t)NN_ * 512 * 4);
    float*          skip  = (float*)bump(p, (size_t)NN_ * 128 * 4);
    float*          hl    = (float*)bump(p, (size_t)NN_ * 128 * 4);
    float*          ha    = (float*)bump(p, (size_t)NN_ * 128 * 4);
    float*          tmpA  = (float*)bump(p, (size_t)NN_ * 128 * 4);
    float*          tmpB  = (float*)bump(p, (size_t)NN_ * 128 * 4);
    float*          tmpC  = (float*)bump(p, (size_t)NN_ * 128 * 4);
    unsigned short* qkvb  = (unsigned short*)bump(p, (size_t)NN_ * 384 * 2);
    float*          sc    = (float*)bump(p, (size_t)GG_ * SS_ * SS_ * 4);
    unsigned short* scb   = (unsigned short*)bump(p, (size_t)GG_ * SS_ * SS_ * 2);
    unsigned short* aob   = (unsigned short*)bump(p, (size_t)NN_ * 128 * 2);
    unsigned short* outb  = (unsigned short*)bump(p, (size_t)NN_ * 128 * 2);
    unsigned short* midb  = (unsigned short*)bump(p, (size_t)NN_ * 256 * 2);
    // bf16 transposed weights [L][Nout][K]
    unsigned short* WqT   = (unsigned short*)bump(p, (size_t)LL_ * 128 * 512 * 2);
    unsigned short* WkT   = (unsigned short*)bump(p, (size_t)LL_ * 128 * 512 * 2);
    unsigned short* WvT   = (unsigned short*)bump(p, (size_t)LL_ * 128 * 512 * 2);
    unsigned short* WeT   = (unsigned short*)bump(p, (size_t)LL_ * 128 * 512 * 2);
    unsigned short* WskT  = (unsigned short*)bump(p, (size_t)LL_ * 128 * 128 * 2);
    unsigned short* WqkvT = (unsigned short*)bump(p, (size_t)LL_ * 128 * 384 * 2);
    unsigned short* WoT   = (unsigned short*)bump(p, (size_t)LL_ * 128 * 128 * 2);
    unsigned short* W1T   = (unsigned short*)bump(p, (size_t)LL_ * 128 * 256 * 2);
    unsigned short* W2T   = (unsigned short*)bump(p, (size_t)LL_ * 256 * 128 * 2);

    cvtT(stream, Wq,   WqT,   128, 512);
    cvtT(stream, Wk,   WkT,   128, 512);
    cvtT(stream, Wv,   WvT,   128, 512);
    cvtT(stream, We,   WeT,   128, 512);
    cvtT(stream, Wsk,  WskT,  128, 128);
    cvtT(stream, Wqkv, WqkvT, 128, 384);
    cvtT(stream, Wo,   WoT,   128, 128);
    cvtT(stream, W1,   W1T,   128, 256);
    cvtT(stream, W2,   W2T,   256, 128);

    atom_encode_kernel<<<(NN_ * 128 + 255) / 256, 256, 0, stream>>>(
        x_feat, atom_emb, pe, Wpe, bpe, x, xb);
    bond_encode_kernel<<<(EE_ * 128 + 255) / 256, 256, 0, stream>>>(
        edge_feat, bond_emb, eb);

    const float inv_sqrt_d = 0.08838834764831845f;  // 1/sqrt(128)
    const long long nND = (long long)NN_ * 128;

    for (int l = 0; l < LL_; ++l) {
        // ---- projections (WMMA, all NT with transposed weights) ----
        launch_gemm(stream, xb, 128, 0, WqT + (size_t)l * 65536, 128, 0,
                    bq + l * 512, q, nullptr, 512, 0, NN_, 512, 128, 1, 0, 1);
        launch_gemm(stream, xb, 128, 0, WkT + (size_t)l * 65536, 128, 0,
                    bk + l * 512, k, nullptr, 512, 0, NN_, 512, 128, 1, 0, 1);
        launch_gemm(stream, xb, 128, 0, WvT + (size_t)l * 65536, 128, 0,
                    bv + l * 512, v, nullptr, 512, 0, NN_, 512, 128, 1, 0, 1);
        launch_gemm(stream, eb, 128, 0, WeT + (size_t)l * 65536, 128, 0,
                    nullptr, nullptr, eebuf, 512, 0, EE_, 512, 128, 1, 0, 1);
        launch_gemm(stream, xb, 128, 0, WskT + (size_t)l * 16384, 128, 0,
                    bsk + l * 128, skip, nullptr, 128, 0, NN_, 128, 128, 1, 0, 1);

        // ---- local TransformerConv (segment softmax over edges) ----
        fill_kernel<<<(NN_ * 4 + 255) / 256, 256, 0, stream>>>(mbuf, -3.402823466e38f,
                                                               (long long)NN_ * 4);
        hipMemsetAsync(zbuf, 0, (size_t)NN_ * 4 * 4, stream);
        hipMemsetAsync(msg, 0, (size_t)NN_ * 512 * 4, stream);
        edge_logits_kernel<<<(EE_ * 4 + 255) / 256, 256, 0, stream>>>(
            edge_idx, q, k, eebuf, logit, mbuf);
        edge_exp_sum_kernel<<<(EE_ * 4 + 255) / 256, 256, 0, stream>>>(
            edge_idx, logit, mbuf, zbuf);
        edge_accum_kernel<<<(unsigned)(((long long)EE_ * 512 + 255) / 256), 256, 0, stream>>>(
            edge_idx, logit, zbuf, v, eebuf, msg);
        combine_local_kernel<<<(unsigned)((nND + 255) / 256), 256, 0, stream>>>(
            msg, skip, x, tmpA);
        bn_kernel<<<128, 256, 0, stream>>>(tmpA, hl, nullptr, g1 + l * 128, b1 + l * 128, NN_);

        // ---- global dense attention per graph ----
        launch_gemm(stream, xb, 128, 0, WqkvT + (size_t)l * 49152, 128, 0,
                    bqkv + l * 384, nullptr, qkvb, 384, 0, NN_, 384, 128, 1, 0, 1);
        launch_gemm(stream, qkvb, 384, (long long)SS_ * 384,
                    qkvb + 128, 384, (long long)SS_ * 384, nullptr,
                    sc, nullptr, 256, (long long)SS_ * SS_,
                    SS_, SS_, 128, 1, 0, GG_);
        softmax_rows_kernel<<<GG_ * SS_, 256, 0, stream>>>(sc, scb, inv_sqrt_d);
        launch_gemm(stream, scb, 256, (long long)SS_ * SS_,
                    qkvb + 256, 384, (long long)SS_ * 384, nullptr,
                    nullptr, aob, 128, (long long)SS_ * 128,
                    SS_, 128, 256, 0, 0, GG_);
        launch_gemm(stream, aob, 128, 0, WoT + (size_t)l * 16384, 128, 0,
                    bo + l * 128, tmpB, nullptr, 128, 0, NN_, 128, 128, 1, 0, 1);
        add2_kernel<<<(unsigned)((nND + 255) / 256), 256, 0, stream>>>(tmpB, x, tmpC, nND);
        bn_kernel<<<128, 256, 0, stream>>>(tmpC, ha, nullptr, g2 + l * 128, b2 + l * 128, NN_);

        // ---- combine + FFN + norm3 ----
        add2_kernel<<<(unsigned)((nND + 255) / 256), 256, 0, stream>>>(hl, ha, tmpA, nND);
        cvt(stream, tmpA, outb, nND);
        launch_gemm(stream, outb, 128, 0, W1T + (size_t)l * 32768, 128, 0,
                    b1m + l * 256, nullptr, midb, 256, 0, NN_, 256, 128, 1, 1, 1);
        launch_gemm(stream, midb, 256, 0, W2T + (size_t)l * 32768, 256, 0,
                    b2m + l * 128, tmpB, nullptr, 128, 0, NN_, 128, 256, 1, 0, 1);
        add2_kernel<<<(unsigned)((nND + 255) / 256), 256, 0, stream>>>(tmpA, tmpB, tmpC, nND);
        bn_kernel<<<128, 256, 0, stream>>>(tmpC, x, xb, g3 + l * 128, b3 + l * 128, NN_);
    }

    head_kernel<<<GG_, 128, 0, stream>>>(x, Wh1, bh1, Wh2, bh2, Wh3, bh3, out);
}
